// LSTM_88716844466216
// MI455X (gfx1250) — matrix operand order repaired and checked by reading, not compiled, not run on hardware
//
#include <hip/hip_runtime.h>
#include <hip/hip_bf16.h>

// ---------------- problem constants ----------------
#define BSZ      64
#define ILEN     1024
#define TLEN     1024
#define TSTEPS   (ILEN + TLEN)
#define IDIM     128
#define HDIM     256
#define GDIM     (4 * HDIM)     // 1024 gate outputs
#define NCLASSES 128
#define NWG      16             // persistent workgroups (one cluster)

typedef __attribute__((ext_vector_type(16))) __bf16 v16bf;
typedef __attribute__((ext_vector_type(8)))  float  v8f;

// ---------------- Tensor Data Mover (TDM) support ----------------
#if __has_builtin(__builtin_amdgcn_tensor_load_to_lds) && __has_builtin(__builtin_amdgcn_s_wait_tensorcnt)
#define HAVE_TDM 1
typedef __attribute__((ext_vector_type(4))) unsigned int v4u_t;
typedef __attribute__((ext_vector_type(8))) int          v8i_t;
typedef __attribute__((ext_vector_type(4))) int          v4i_t;

// Issue a 1-D TDM copy of `nelem` bf16 elements from global to LDS.
// D# packing per CDNA5 ISA §8.3 (group0) / §8.4 (group1); groups 2/3 unused (2-D max).
__device__ __forceinline__ void tdm_load_1d(const void* gptr, void* lptr, unsigned nelem) {
    const unsigned long long ga  = (unsigned long long)gptr;
    const unsigned           lds = (unsigned)(unsigned long long)(size_t)lptr;
    v4u_t g0;
    g0[0] = 1u;                                                   // [1:0] count = 1
    g0[1] = lds;                                                  // [63:32] lds_addr
    g0[2] = (unsigned)(ga & 0xffffffffu);                         // [95:64] global_addr lo
    g0[3] = (unsigned)((ga >> 32) & 0x01ffffffu) | (2u << 30);    // [120:96] addr hi, [127:126] type=2
    v8i_t g1;
    g1[0] = (1 << 16);                                            // [17:16] data_size = 1 (2 bytes)
    g1[1] = (int)((nelem & 0xffffu) << 16);                       // tensor_dim0[15:0] @ [63:48]
    g1[2] = (int)((nelem >> 16) & 0xffffu) | (1 << 16);           // tensor_dim0[31:16]; tensor_dim1 = 1
    g1[3] = (int)((nelem & 0xffffu) << 16);                       // tile_dim0 = nelem @ [127:112]
    g1[4] = 1;                                                    // tile_dim1 = 1, tile_dim2 = 0
    g1[5] = (int)nelem;                                           // tensor_dim0_stride[31:0]
    g1[6] = 0;                                                    // stride hi / dim1 stride
    g1[7] = 0;
    v4i_t z4 = {0, 0, 0, 0};
#if defined(__clang_major__) && (__clang_major__ >= 23)
    v8i_t z8 = {0, 0, 0, 0, 0, 0, 0, 0};
    __builtin_amdgcn_tensor_load_to_lds(g0, g1, z4, z4, z8, 0);   // 6-arg (clang-23 headers)
#else
    __builtin_amdgcn_tensor_load_to_lds(g0, g1, z4, z4, 0);       // 5-arg (ROCm 7.2 clang-22)
#endif
}
#else
#define HAVE_TDM 0
#endif

// ---------------- WMMA fragment loaders (per ISA 16-bit layouts) ----------------
// A (16x32, MxK): lanes 0-15 -> M=lane, elems 0-7 = K k0..k0+7, elems 8-15 = K k0+16..k0+23
//                 lanes 16-31 -> M=lane-16, K ranges shifted by +8
__device__ __forceinline__ v16bf ldA(const __bf16* base, int ld, int k0, int lane) {
    const int m  = lane & 15;
    const int kh = lane >> 4;
    const __bf16* row = base + m * ld + k0 + kh * 8;
    union { uint4 q[2]; v16bf v; } u;
    u.q[0] = *reinterpret_cast<const uint4*>(row);
    u.q[1] = *reinterpret_cast<const uint4*>(row + 16);
    return u.v;
}
// B (32x16, KxN): B[k][n] = W[n0+n][k] (row-major weights). lane n<16 holds K k0..k0+7, k0+16..k0+23,
// lane 16+n holds K k0+8..k0+15, k0+24..k0+31 of column n.
__device__ __forceinline__ v16bf ldB(const __bf16* wbase, int ld, int k0, int lane) {
    const int n  = lane & 15;
    const int kh = lane >> 4;
    const __bf16* row = wbase + n * ld + k0 + kh * 8;
    union { uint4 q[2]; v16bf v; } u;
    u.q[0] = *reinterpret_cast<const uint4*>(row);
    u.q[1] = *reinterpret_cast<const uint4*>(row + 16);
    return u.v;
}

__device__ __forceinline__ float fsig(float x) { return 1.0f / (1.0f + __expf(-x)); }

// ---------------- prep: transpose input to [t][b][k] and cast to bf16 ----------------
__global__ void k_prep(const float* __restrict__ inp, unsigned short* __restrict__ xraw) {
    __bf16* x = reinterpret_cast<__bf16*>(xraw);
    const size_t total  = (size_t)BSZ * ILEN * IDIM;
    const size_t stride = (size_t)gridDim.x * blockDim.x;
    for (size_t d = (size_t)blockIdx.x * blockDim.x + threadIdx.x; d < total; d += stride) {
        size_t t = d / (BSZ * IDIM);
        size_t r = d - t * (BSZ * IDIM);
        size_t b = r >> 7;        // /IDIM
        size_t k = r & (IDIM - 1);
        x[d] = (__bf16)inp[(b * ILEN + t) * IDIM + k];
    }
}

// ---------------- init: zero h ping-pong, cast clf_w, reset barrier ----------------
__global__ void k_init(unsigned short* __restrict__ hraw, unsigned short* __restrict__ cwraw,
                       const float* __restrict__ cw, unsigned int* __restrict__ bar) {
    __bf16* hping = reinterpret_cast<__bf16*>(hraw);
    __bf16* cwbf  = reinterpret_cast<__bf16*>(cwraw);
    const int i = blockIdx.x * blockDim.x + threadIdx.x;    // 16384 threads
    hping[i]         = (__bf16)0.0f;
    hping[i + 16384] = (__bf16)0.0f;
    cwbf[i]          = (__bf16)cw[i];
    cwbf[i + 16384]  = (__bf16)cw[i + 16384];
    if (i == 0) *bar = 0u;
}

// ---------------- persistent recurrent kernel ----------------
// 16 WGs x 256 threads. WG wg owns hidden units [wg*16, wg*16+16): the 4 matching
// gate rows of w_ih/w_hh live in LDS as bf16, c-state stays in LDS forever.
__global__ void __launch_bounds__(256, 1)
k_lstm(const unsigned short* __restrict__ xraw,
       const float* __restrict__ w_ih, const float* __restrict__ w_hh,
       const float* __restrict__ bias,
       unsigned short* __restrict__ hraw, unsigned short* __restrict__ hsraw,
       unsigned int* __restrict__ bar) {
    const __bf16* x  = reinterpret_cast<const __bf16*>(xraw);
    __bf16* hping    = reinterpret_cast<__bf16*>(hraw);
    __bf16* hs       = reinterpret_cast<__bf16*>(hsraw);

    extern __shared__ char smem[];
    __bf16* whh_l = (__bf16*)(smem);                    // 64 x 256  (32 KB)
    __bf16* wih_l = (__bf16*)(smem + 32768);            // 64 x 128  (16 KB)
    __bf16* hbuf  = (__bf16*)(smem + 49152);            // 64 x 256  (32 KB)
    __bf16* xbuf  = (__bf16*)(smem + 81920);            // 64 x 128  (16 KB)
    float*  zbuf  = (float*)(smem + 98304);             // 64 x 64   (16 KB)
    float*  cbuf  = (float*)(smem + 114688);            // 64 x 16   ( 4 KB)
    float*  bl    = (float*)(smem + 118784);            // 64        (256 B)

    const int tid  = threadIdx.x;
    const int wg   = blockIdx.x;      // 0..15
    const int lane = tid & 31;
    const int wid  = tid >> 5;        // 0..7

    // ---- one-time: stage this WG's weight slice into LDS as bf16 ----
    for (int e = tid; e < 64 * HDIM; e += 256) {
        int r = e >> 8, k = e & 255;
        int j = (r >> 4) * HDIM + wg * 16 + (r & 15);   // gate-row index in [0,1024)
        whh_l[e] = (__bf16)w_hh[j * HDIM + k];
    }
    for (int e = tid; e < 64 * IDIM; e += 256) {
        int r = e >> 7, k = e & 127;
        int j = (r >> 4) * HDIM + wg * 16 + (r & 15);
        wih_l[e] = (__bf16)w_ih[j * IDIM + k];
    }
    if (tid < 64) {
        int j = (tid >> 4) * HDIM + wg * 16 + (tid & 15);
        bl[tid] = bias[j];
    }
    for (int e = tid; e < BSZ * 16; e += 256) cbuf[e] = 0.0f;
    __syncthreads();

    for (int t = 0; t < TSTEPS; ++t) {
        const __bf16* hr = hping + (size_t)(t & 1) * (BSZ * HDIM);
        __bf16*       hw = hping + (size_t)((t + 1) & 1) * (BSZ * HDIM);
        const bool hasx = (t < ILEN);

#if HAVE_TDM
        // ---- DMA the h vector (32 KB) and x tile (16 KB) into LDS via the TDM ----
        if (wid == 0) {
            tdm_load_1d(hr, hbuf, BSZ * HDIM);
            if (hasx) tdm_load_1d(x + (size_t)t * BSZ * IDIM, xbuf, BSZ * IDIM);
            __builtin_amdgcn_s_wait_tensorcnt(0);
        }
        if (hasx && (t + 1 < ILEN)) {  // warm L2 for next timestep's x tile
            const char* nx = (const char*)(x + (size_t)(t + 1) * BSZ * IDIM);
            __builtin_prefetch(nx + tid * 64, 0, 1);
        }
#else
        // ---- fallback: stage via VGPR b128 copies ----
        {
            const uint4* src = (const uint4*)hr;
            uint4*       dst = (uint4*)hbuf;
            for (int e = tid; e < (BSZ * HDIM) / 8; e += 256) dst[e] = src[e];
        }
        if (hasx) {
            const uint4* src = (const uint4*)(x + (size_t)t * BSZ * IDIM);
            uint4*       dst = (uint4*)xbuf;
            for (int e = tid; e < (BSZ * IDIM) / 8; e += 256) dst[e] = src[e];
            if (t + 1 < ILEN) {
                const char* nx = (const char*)(x + (size_t)(t + 1) * BSZ * IDIM);
                __builtin_prefetch(nx + tid * 64, 0, 1);
            }
        }
#endif
        __syncthreads();

        // ---- WMMA: each wave computes 2 of the WG's 16 z-tiles (16x16) ----
#pragma unroll
        for (int j = 0; j < 2; ++j) {
            const int tile = wid * 2 + j;
            const int rt = tile >> 2;      // batch row-tile 0..3
            const int gi = tile & 3;       // gate index (i,f,g,o)
            const float bv = bl[gi * 16 + (lane & 15)];
            v8f acc = {bv, bv, bv, bv, bv, bv, bv, bv};

            const __bf16* abase = hbuf + rt * 16 * HDIM;
            const __bf16* bbase = whh_l + gi * 16 * HDIM;
#pragma unroll
            for (int k0 = 0; k0 < HDIM; k0 += 32) {
                v16bf a = ldA(abase, HDIM, k0, lane);
                v16bf b = ldB(bbase, HDIM, k0, lane);
                acc = __builtin_amdgcn_wmma_f32_16x16x32_bf16(
                    false, a, false, b, (short)0, acc, false, false);
            }
            if (hasx) {
                const __bf16* a2 = xbuf + rt * 16 * IDIM;
                const __bf16* b2 = wih_l + gi * 16 * IDIM;
#pragma unroll
                for (int k0 = 0; k0 < IDIM; k0 += 32) {
                    v16bf a = ldA(a2, IDIM, k0, lane);
                    v16bf b = ldB(b2, IDIM, k0, lane);
                    acc = __builtin_amdgcn_wmma_f32_16x16x32_bf16(
                        false, a, false, b, (short)0, acc, false, false);
                }
            }
            // C/D layout: VGPR r -> M=r (lanes 0-15) or M=8+r (lanes 16-31), N=lane&15
            const int n  = lane & 15;
            const int mh = (lane < 16) ? 0 : 8;
#pragma unroll
            for (int r = 0; r < 8; ++r)
                zbuf[(rt * 16 + mh + r) * 64 + gi * 16 + n] = acc[r];
        }
        __syncthreads();

        // ---- elementwise gates + state update (c stays in LDS) ----
#pragma unroll
        for (int e = 0; e < 4; ++e) {
            const int idx = tid * 4 + e;            // 64 batch x 16 hidden
            const int b  = idx >> 4;
            const int kl = idx & 15;
            const float* zrow = zbuf + b * 64;
            const float ig = fsig(zrow[kl]);
            const float fg = fsig(zrow[16 + kl]);
            const float gg = tanhf(zrow[32 + kl]);
            const float og = fsig(zrow[48 + kl]);
            const float cn = fg * cbuf[idx] + ig * gg;
            cbuf[idx] = cn;
            const float hv = og * tanhf(cn);
            const __bf16 hb = (__bf16)hv;
            hw[b * HDIM + wg * 16 + kl] = hb;
            if (t >= ILEN)
                hs[((size_t)b * TLEN + (t - ILEN)) * HDIM + wg * 16 + kl] = hb;
        }

        // ---- one grid barrier per step (ping-pong h makes a single sync safe) ----
        __threadfence();
        __syncthreads();
#if __has_builtin(__builtin_amdgcn_s_cluster_barrier)
        __builtin_amdgcn_s_cluster_barrier();       // cluster-dispatch fast path
#endif
        if (tid == 0) {                             // portable atomic fallback barrier
            __atomic_fetch_add(bar, 1u, __ATOMIC_RELEASE);
            const unsigned target = (unsigned)(t + 1) * NWG;
            while (__atomic_load_n(bar, __ATOMIC_ACQUIRE) < target) {
#if __has_builtin(__builtin_amdgcn_s_sleep)
                __builtin_amdgcn_s_sleep(2);
#endif
            }
        }
        __syncthreads();
    }
}

// ---------------- classifier GEMM: out[65536,128] = hs[65536,256] @ clf_w^T + clf_b ----
__global__ void __launch_bounds__(256)
k_clf(const unsigned short* __restrict__ hsraw, const unsigned short* __restrict__ cwraw,
      const float* __restrict__ cb, float* __restrict__ out) {
    const __bf16* hs = reinterpret_cast<const __bf16*>(hsraw);
    const __bf16* cw = reinterpret_cast<const __bf16*>(cwraw);
    const int wid  = threadIdx.x >> 5;
    const int lane = threadIdx.x & 31;
    const int tile = blockIdx.x * 8 + wid;          // 32768 tiles, 1 per wave
    const int rowTile = tile >> 3;
    const int n0 = (tile & 7) * 16;
    const size_t rowBase = (size_t)rowTile * 16;

    const float bv = cb[n0 + (lane & 15)];
    v8f acc = {bv, bv, bv, bv, bv, bv, bv, bv};

    const __bf16* abase = hs + rowBase * HDIM;
    const __bf16* bbase = cw + (size_t)n0 * HDIM;
#pragma unroll
    for (int k0 = 0; k0 < HDIM; k0 += 32) {
        v16bf a = ldA(abase, HDIM, k0, lane);
        v16bf b = ldB(bbase, HDIM, k0, lane);
        acc = __builtin_amdgcn_wmma_f32_16x16x32_bf16(
            false, a, false, b, (short)0, acc, false, false);
    }
    const int n  = lane & 15;
    const int mh = (lane < 16) ? 0 : 8;
#pragma unroll
    for (int r = 0; r < 8; ++r)
        out[(rowBase + mh + r) * NCLASSES + n0 + n] = acc[r];
}

// ---------------- host launch ----------------
extern "C" void kernel_launch(void* const* d_in, const int* in_sizes, int n_in,
                              void* d_out, int out_size, void* d_ws, size_t ws_size,
                              hipStream_t stream) {
    const float* inp   = (const float*)d_in[0];
    // d_in[1] = tlen scalar (compile-time constant here)
    const float* w_ih  = (const float*)d_in[2];
    const float* w_hh  = (const float*)d_in[3];
    const float* bias  = (const float*)d_in[4];
    const float* clf_w = (const float*)d_in[5];
    const float* clf_b = (const float*)d_in[6];
    float* out = (float*)d_out;
    char*  ws  = (char*)d_ws;

    // workspace layout (bytes), 256B aligned
    size_t off = 0;
    auto take = [&](size_t bytes) { size_t o = off; off = (off + bytes + 255) & ~(size_t)255; return o; };
    unsigned short* x_bf  = (unsigned short*)(ws + take((size_t)BSZ * ILEN * IDIM * 2));  // 16 MB
    unsigned short* hping = (unsigned short*)(ws + take((size_t)2 * BSZ * HDIM * 2));     // 64 KB
    unsigned short* cw_bf = (unsigned short*)(ws + take((size_t)NCLASSES * HDIM * 2));    // 64 KB
    unsigned int*   bar   = (unsigned int*)  (ws + take(256));
    unsigned short* hs_bf = (unsigned short*)(ws + take((size_t)BSZ * TLEN * HDIM * 2));  // 32 MB
    (void)ws_size; (void)in_sizes; (void)n_in; (void)out_size;

    k_prep<<<2048, 256, 0, stream>>>(inp, x_bf);
    k_init<<<64, 256, 0, stream>>>(hping, cw_bf, clf_w, bar);

    const size_t shmem = 119040;  // 116.25 KB of the 320 KB/WGP LDS
    k_lstm<<<NWG, 256, shmem, stream>>>(x_bf, w_ih, w_hh, bias, hping, hs_bf, bar);

    k_clf<<<4096, 256, 0, stream>>>(hs_bf, cw_bf, clf_b, out);
}
